// MultimodalHypergraphAnomalyDetector_59158879535698
// MI455X (gfx1250) — matrix operand order repaired and checked by reading, not compiled, not run on hardware
//
#include <hip/hip_runtime.h>
#include <hip/hip_bf16.h>
#include <math.h>

// ---------------------------------------------------------------------------
// Types
// ---------------------------------------------------------------------------
typedef __bf16 bf16_t;
typedef bf16_t v16bf __attribute__((ext_vector_type(16)));
typedef float  v8f   __attribute__((ext_vector_type(8)));
typedef float  f4    __attribute__((ext_vector_type(4)));
typedef unsigned short u16x8 __attribute__((ext_vector_type(8)));

__device__ __forceinline__ unsigned short f2bf(float f) {
  bf16_t h = (bf16_t)f;
  return __builtin_bit_cast(unsigned short, h);
}

#define NB 3072
#define TOPK_ 16

// ---------------------------------------------------------------------------
// Generic WMMA GEMM:  C = act(alpha * A@B(^T) [+ C_old] [+ bias)]
// A: [M,K] row-major (lda=K).  abf=1 -> A is bf16 (async-copied straight into
// LDS via GLOBAL_LOAD_ASYNC_TO_LDS_B128); abf=0 -> A is fp32 (convert in LDS).
// B: bt=0 -> [K,N] fp32 (ldb), bt=1 -> [N,K] fp32 (ldb=K).
// 128x128 tile per block, 256 threads = 8 waves, each wave 64x32 (4x2 wmma).
// ---------------------------------------------------------------------------
#define TBM 128
#define TBN 128
#define TBK 32
#define LDSP 40  // padded k-stride (elements); row stride = 80B (16B aligned)

union Frag { v16bf v; u16x8 h[2]; };

__global__ __launch_bounds__(256)
void gemm_bf16_wmma(const void* __restrict__ Av, const float* __restrict__ Bm,
                    const float* __restrict__ bias, float* __restrict__ C,
                    int M, int N, int K, int lda, int ldb, int ldc,
                    float alpha, int bt, int act, int hasBias, int accIn,
                    int abf)
{
  __shared__ unsigned short sA[TBM * LDSP];
  __shared__ unsigned short sB[TBN * LDSP];

  const int tid  = threadIdx.x;
  const int lane = tid & 31;
  const int wid  = tid >> 5;
  const int lm   = lane & 15;
  const int lh   = lane >> 4;
  const int wr   = (wid & 1) * 64;   // wave row base within tile
  const int wc   = (wid >> 1) * 32;  // wave col base within tile

  const int bm = blockIdx.y * TBM;
  const int bn = blockIdx.x * TBN;

  // low 32 bits of the generic shared pointer == LDS byte offset (ISA 10.2)
  const unsigned sAbase = (unsigned)(uintptr_t)&sA[0];

  v8f acc[4][2];
  for (int i = 0; i < 4; i++)
    for (int j = 0; j < 2; j++)
      for (int e = 0; e < 8; e++) acc[i][j][e] = 0.0f;

  for (int k0 = 0; k0 < K; k0 += TBK) {
    bool asyncUsed = false;
    // ------------------------------------------------------------------ A
    if (abf) {
      const unsigned short* Ab = (const unsigned short*)Av;
      if (bm + TBM <= M && k0 + TBK <= K) {
        // full tile: 128x32 bf16 = 512 x 16B chunks; 2 per thread, async DMA
        const unsigned short* gbase = Ab + (size_t)bm * lda + k0; // uniform
        asyncUsed = true;
#pragma unroll
        for (int qq = 0; qq < 2; qq++) {
          int q  = tid + qq * 256;
          int r  = q >> 2;          // row in tile
          int kc = (q & 3) << 3;    // k chunk: 0,8,16,24
          unsigned goff  = (unsigned)((r * lda + kc) * 2);
          unsigned laddr = sAbase + (unsigned)((r * LDSP + kc) * 2);
          asm volatile("global_load_async_to_lds_b128 %0, %1, %2"
                       :: "v"(laddr), "v"(goff), "s"(gbase) : "memory");
        }
      } else {
        for (int q = tid; q < (TBM * TBK) / 8; q += 256) {
          int r  = q >> 2;
          int kc = (q & 3) << 3;
          int gr = bm + r, gk = k0 + kc;
          u16x8 v = (u16x8)0;
          if (gr < M && gk + 7 < K)
            v = *(const u16x8*)(Ab + (size_t)gr * lda + gk);
          *(u16x8*)&sA[r * LDSP + kc] = v;
        }
      }
    } else {
      const float* A = (const float*)Av;
      for (int q = tid; q < (TBM * TBK) / 4; q += 256) {
        int r  = q >> 3;
        int kq = (q & 7) << 2;
        int gr = bm + r, gk = k0 + kq;
        float x0 = 0, x1 = 0, x2 = 0, x3 = 0;
        if (gr < M) {
          const float* ap = A + (size_t)gr * lda + gk;
          if (gk + 3 < K) {
            f4 v = *(const f4*)ap; x0 = v.x; x1 = v.y; x2 = v.z; x3 = v.w;
            if (gk + TBK < K) __builtin_prefetch(ap + TBK, 0, 1);
          } else {
            if (gk + 0 < K) x0 = ap[0];
            if (gk + 1 < K) x1 = ap[1];
            if (gk + 2 < K) x2 = ap[2];
            if (gk + 3 < K) x3 = ap[3];
          }
        }
        unsigned short* d = &sA[r * LDSP + kq];
        d[0] = f2bf(x0); d[1] = f2bf(x1); d[2] = f2bf(x2); d[3] = f2bf(x3);
      }
    }
    // ------------------------------------------------------------------ B
    if (bt) {  // B is [N,K]: rows copy straight (already k-contiguous)
      for (int q = tid; q < (TBN * TBK) / 4; q += 256) {
        int r  = q >> 3;
        int kq = (q & 7) << 2;
        int gn = bn + r, gk = k0 + kq;
        float x0 = 0, x1 = 0, x2 = 0, x3 = 0;
        if (gn < N) {
          const float* bp = Bm + (size_t)gn * ldb + gk;
          if (gk + 3 < K) {
            f4 v = *(const f4*)bp; x0 = v.x; x1 = v.y; x2 = v.z; x3 = v.w;
            if (gk + TBK < K) __builtin_prefetch(bp + TBK, 0, 1);
          } else {
            if (gk + 0 < K) x0 = bp[0];
            if (gk + 1 < K) x1 = bp[1];
            if (gk + 2 < K) x2 = bp[2];
            if (gk + 3 < K) x3 = bp[3];
          }
        }
        unsigned short* d = &sB[r * LDSP + kq];
        d[0] = f2bf(x0); d[1] = f2bf(x1); d[2] = f2bf(x2); d[3] = f2bf(x3);
      }
    } else {   // B is [K,N]: coalesced read along N, transposed scatter to LDS
      for (int q = tid; q < (TBK * TBN) / 4; q += 256) {
        int kk = q >> 5;
        int n4 = (q & 31) << 2;
        int gk = k0 + kk, gn = bn + n4;
        float x0 = 0, x1 = 0, x2 = 0, x3 = 0;
        if (gk < K) {
          const float* bp = Bm + (size_t)gk * ldb + gn;
          if (gn + 3 < N) {
            f4 v = *(const f4*)bp; x0 = v.x; x1 = v.y; x2 = v.z; x3 = v.w;
            if (gk + TBK < K) __builtin_prefetch(bp + (size_t)TBK * ldb, 0, 1);
          } else {
            if (gn + 0 < N) x0 = bp[0];
            if (gn + 1 < N) x1 = bp[1];
            if (gn + 2 < N) x2 = bp[2];
            if (gn + 3 < N) x3 = bp[3];
          }
        }
        sB[(n4 + 0) * LDSP + kk] = f2bf(x0);
        sB[(n4 + 1) * LDSP + kk] = f2bf(x1);
        sB[(n4 + 2) * LDSP + kk] = f2bf(x2);
        sB[(n4 + 3) * LDSP + kk] = f2bf(x3);
      }
    }
    if (asyncUsed)
      asm volatile("s_wait_asynccnt 0x0" ::: "memory");
    __syncthreads();

    // ---- fragments per ISA VGPR layouts (wave32)
    Frag afrag[4], bfrag[2];
#pragma unroll
    for (int mt = 0; mt < 4; mt++) {
      const unsigned short* base = &sA[(wr + mt * 16 + lm) * LDSP];
      afrag[mt].h[0] = *(const u16x8*)(base + 8 * lh);        // K = 8h..8h+7
      afrag[mt].h[1] = *(const u16x8*)(base + 16 + 8 * lh);   // K = 16+8h..+7
    }
#pragma unroll
    for (int nt = 0; nt < 2; nt++) {
      const unsigned short* base = &sB[(wc + nt * 16 + lm) * LDSP];
      bfrag[nt].h[0] = *(const u16x8*)(base + 8 * lh);        // K = 8h..8h+7
      bfrag[nt].h[1] = *(const u16x8*)(base + 16 + 8 * lh);   // K = 16+8h..+7
    }
#pragma unroll
    for (int mt = 0; mt < 4; mt++)
#pragma unroll
      for (int nt = 0; nt < 2; nt++)
        acc[mt][nt] = __builtin_amdgcn_wmma_f32_16x16x32_bf16(
            false, afrag[mt].v, false, bfrag[nt].v, (short)0, acc[mt][nt],
            false, false);
    __syncthreads();
  }

  // ---- epilogue (C/D layout: m = lh*8 + vgpr, n = lane&15)
  for (int mt = 0; mt < 4; mt++) {
    for (int nt = 0; nt < 2; nt++) {
      int col = bn + wc + nt * 16 + lm;
      if (col >= N) continue;
      float bv = hasBias ? bias[col] : 0.0f;
#pragma unroll
      for (int e = 0; e < 8; e++) {
        int row = bm + wr + mt * 16 + lh * 8 + e;
        if (row >= M) continue;
        size_t off = (size_t)row * ldc + col;
        float v = alpha * acc[mt][nt][e];
        if (accIn) v += C[off];
        v += bv;
        if (act == 1)      v = fmaxf(v, 0.0f);
        else if (act == 2) v = tanhf(v);
        else if (act == 3) v = 1.0f / (1.0f + expf(-v));
        C[off] = v;
      }
    }
  }
}

// ---------------------------------------------------------------------------
// Elementwise / reduction helpers
// ---------------------------------------------------------------------------
__global__ void fill_kernel(float* p, float v, size_t n) {
  size_t i = (size_t)blockIdx.x * 256 + threadIdx.x;
  if (i < n) p[i] = v;
}

__global__ void pool_kernel(const float* __restrict__ img, float* __restrict__ p, int B) {
  int idx = blockIdx.x * 256 + threadIdx.x;
  if (idx >= B * 192) return;
  int b = idx / 192, f = idx % 192;
  int c = f >> 6, rem = f & 63;
  int oy = rem >> 3, ox = rem & 7;
  const float* base = img + (((size_t)b * 3 + c) * 32 + oy * 4) * 32 + ox * 4;
  float s = 0.0f;
  for (int iy = 0; iy < 4; iy++)
    for (int ix = 0; ix < 4; ix++) s += base[iy * 32 + ix];
  p[idx] = s * (1.0f / 16.0f);
}

// BatchNorm over rows for one column per block; writes into dst[:, dcol+j]
__global__ void bn_kernel(const float* __restrict__ src, const float* __restrict__ g,
                          const float* __restrict__ bt, float* __restrict__ dst,
                          int M, int lds, int ldd, int dcol) {
  int j = blockIdx.x, tid = threadIdx.x;
  __shared__ float s1[256], s2[256];
  float a = 0, b = 0;
  for (int i = tid; i < M; i += 256) {
    float x = src[(size_t)i * lds + j];
    a += x; b += x * x;
  }
  s1[tid] = a; s2[tid] = b; __syncthreads();
  for (int st = 128; st > 0; st >>= 1) {
    if (tid < st) { s1[tid] += s1[tid + st]; s2[tid] += s2[tid + st]; }
    __syncthreads();
  }
  float m  = s1[0] / M;
  float v  = s2[0] / M - m * m;
  float rs = rsqrtf(v + 1e-5f);
  float gg = g[j], bb = bt[j];
  for (int i = tid; i < M; i += 256)
    dst[(size_t)i * ldd + dcol + j] = (src[(size_t)i * lds + j] - m) * rs * gg + bb;
}

__global__ void rownorm_kernel(float* z, int N) {  // block per row, blockDim==N==256
  int row = blockIdx.x, tid = threadIdx.x;
  __shared__ float s[256];
  float x = z[(size_t)row * N + tid];
  s[tid] = x * x; __syncthreads();
  for (int st = 128; st > 0; st >>= 1) { if (tid < st) s[tid] += s[tid + st]; __syncthreads(); }
  z[(size_t)row * N + tid] = x / (sqrtf(s[0]) + 1e-8f);
}

// top-16 per sim row -> bf16 H, H^T, degree accumulators (atomic), De
__global__ __launch_bounds__(256)
void topk_kernel(const float* __restrict__ sim, unsigned short* __restrict__ Hm,
                 unsigned short* __restrict__ HT, float* __restrict__ Dv,
                 float* __restrict__ Dv2, float* __restrict__ De) {
  __shared__ float sv[NB];
  __shared__ float rv[256];
  __shared__ int   ri[256];
  __shared__ float hvs[TOPK_];
  __shared__ int   idxs[TOPK_];
  __shared__ float sew;
  int row = blockIdx.x, tid = threadIdx.x;
  for (int i = tid; i < NB; i += 256) sv[i] = sim[(size_t)row * NB + i];
  __syncthreads();
  for (int t = 0; t < TOPK_; t++) {
    float bv = -3.0e38f; int bi = 0;
    for (int i = tid; i < NB; i += 256) { float x = sv[i]; if (x > bv) { bv = x; bi = i; } }
    rv[tid] = bv; ri[tid] = bi; __syncthreads();
    for (int st = 128; st > 0; st >>= 1) {
      if (tid < st && rv[tid + st] > rv[tid]) { rv[tid] = rv[tid + st]; ri[tid] = ri[tid + st]; }
      __syncthreads();
    }
    if (tid == 0) { idxs[t] = ri[0]; hvs[t] = rv[0]; sv[ri[0]] = -3.0e38f; }
    __syncthreads();
  }
  if (tid == 0) {
    float sum = 0.0f;
    for (int t = 0; t < TOPK_; t++) {
      float hv = 1.0f / (1.0f + expf(-hvs[t]));
      if (!(hv > 0.5f)) hv = 0.0f;
      hvs[t] = hv; sum += hv;
    }
    De[row] = sum + 1e-8f;
    sew = sum / (float)TOPK_;
  }
  __syncthreads();
  if (tid < TOPK_) {
    float hv = hvs[tid]; int id = idxs[tid];
    unsigned short hb = f2bf(hv);
    Hm[(size_t)id * NB + row] = hb;   // H[idx, row]
    HT[(size_t)row * NB + id] = hb;   // H^T[row, idx]
    atomicAdd(&Dv[id],  hv * sew);
    atomicAdd(&Dv2[id], hv);
  }
}

__global__ void prep_kernel(const float* Dv, const float* De, float* dvis, float* ide, int n) {
  int i = blockIdx.x * 256 + threadIdx.x;
  if (i < n) { dvis[i] = rsqrtf(Dv[i]); ide[i] = 1.0f / De[i]; }
}

__global__ void scale_rows_kernel(const float* src, const float* s, float c,
                                  float* dst, int M, int N) {
  size_t idx = (size_t)blockIdx.x * 256 + threadIdx.x;
  if (idx < (size_t)M * N) dst[idx] = c * s[idx / N] * src[idx];
}

__global__ void axpby_kernel(float a, const float* x, float b, const float* y,
                             float* dst, size_t n) {
  size_t i = (size_t)blockIdx.x * 256 + threadIdx.x;
  if (i < n) dst[i] = a * x[i] + b * y[i];
}

__global__ void sqdiff_kernel(const float* a, const float* b, float* out, size_t n) {
  __shared__ float s[256];
  size_t i = (size_t)blockIdx.x * 256 + threadIdx.x;
  float d = 0.0f;
  if (i < n) { float t = a[i] - b[i]; d = t * t; }
  s[threadIdx.x] = d; __syncthreads();
  for (int st = 128; st > 0; st >>= 1) {
    if (threadIdx.x < st) s[threadIdx.x] += s[threadIdx.x + st];
    __syncthreads();
  }
  if (threadIdx.x == 0) atomicAdd(out, s[0]);
}

__global__ void speccols_kernel(const float* r, const float* t2, const float* Dv2,
                                float* num, float* den, int M, int C) {
  int c = blockIdx.x, tid = threadIdx.x;
  __shared__ float s1[256], s2[256];
  float a = 0, b = 0;
  for (int i = tid; i < M; i += 256) {
    float rvv = r[(size_t)i * C + c];
    float lc  = Dv2[i] * rvv - t2[(size_t)i * C + c];
    a += rvv * lc;
    b += Dv2[i] * rvv * rvv;
  }
  s1[tid] = a; s2[tid] = b; __syncthreads();
  for (int st = 128; st > 0; st >>= 1) {
    if (tid < st) { s1[tid] += s1[tid + st]; s2[tid] += s2[tid + st]; }
    __syncthreads();
  }
  if (tid == 0) { num[c] = s1[0]; den[c] = s2[0]; }
}

__global__ void finalize_kernel(const float* rsum, const float* num, const float* den,
                                float* out, float inv_count, int C) {
  __shared__ float s[64];
  int tid = threadIdx.x;
  float v = (tid < C) ? num[tid] / (den[tid] + 1e-8f) : 0.0f;
  s[tid] = v; __syncthreads();
  for (int st = 32; st > 0; st >>= 1) { if (tid < st) s[tid] += s[tid + st]; __syncthreads(); }
  if (tid == 0) {
    out[0] = rsum[0] * inv_count;  // recon loss
    out[1] = s[0] / (float)C;      // spectral loss
  }
}

// ---------------------------------------------------------------------------
// Host orchestration
// ---------------------------------------------------------------------------
static inline void launch_gemm(hipStream_t s, const void* A, const float* Bm,
                               const float* bias, float* C, int M, int N, int K,
                               int ldb, int ldc, float alpha, int bt, int act,
                               int hasBias, int accIn, int abf) {
  dim3 g((unsigned)((N + TBN - 1) / TBN), (unsigned)((M + TBM - 1) / TBM));
  gemm_bf16_wmma<<<g, 256, 0, s>>>(A, Bm, bias, C, M, N, K, K, ldb, ldc,
                                   alpha, bt, act, hasBias, accIn, abf);
}

extern "C" void kernel_launch(void* const* d_in, const int* in_sizes, int n_in,
                              void* d_out, int out_size, void* d_ws, size_t ws_size,
                              hipStream_t stream) {
  const int B = NB;
  const float* images = (const float*)d_in[0];
  const float* txt    = (const float*)d_in[1];
  const float* sig    = (const float*)d_in[2];
  const float* iw1 = (const float*)d_in[3];  const float* ib1 = (const float*)d_in[4];
  const float* iw2 = (const float*)d_in[5];  const float* ib2 = (const float*)d_in[6];
  const float* ig  = (const float*)d_in[7];  const float* ibt = (const float*)d_in[8];
  const float* tw1 = (const float*)d_in[9];  const float* tb1 = (const float*)d_in[10];
  const float* tw2 = (const float*)d_in[11]; const float* tb2 = (const float*)d_in[12];
  const float* tg  = (const float*)d_in[13]; const float* tbt = (const float*)d_in[14];
  const float* sw1 = (const float*)d_in[15]; const float* sb1 = (const float*)d_in[16];
  const float* sw2 = (const float*)d_in[17]; const float* sb2 = (const float*)d_in[18];
  const float* sg  = (const float*)d_in[19]; const float* sbt = (const float*)d_in[20];
  const float* gw  = (const float*)d_in[21]; const float* gb  = (const float*)d_in[22];
  const float* c1w = (const float*)d_in[23]; const float* c1b = (const float*)d_in[24];
  const float* c2w = (const float*)d_in[25]; const float* c2b = (const float*)d_in[26];
  const float* rw1 = (const float*)d_in[27]; const float* rb1 = (const float*)d_in[28];
  const float* rw2 = (const float*)d_in[29]; const float* rb2 = (const float*)d_in[30];
  const float* aw1 = (const float*)d_in[31]; const float* ab1 = (const float*)d_in[32];
  const float* aw2 = (const float*)d_in[33]; const float* ab2 = (const float*)d_in[34];
  const float* dw1 = (const float*)d_in[35]; const float* db1 = (const float*)d_in[36];
  const float* dw2 = (const float*)d_in[37]; const float* db2 = (const float*)d_in[38];
  float* out = (float*)d_out;

  // workspace carve-up (floats; all offsets multiples of 4 -> 16B aligned)
  float* w = (float*)d_ws;
  auto take = [&](size_t n) { float* p = w; w += n; return p; };
  size_t bb = (size_t)B * B;
  float* POOL = take((size_t)B * 192);
  float* MH   = take((size_t)B * 256);
  float* MY   = take((size_t)B * 256);
  float* FE   = take((size_t)B * 768);
  float* Z    = take((size_t)B * 256);
  float* SIM  = take(bb);
  unsigned short* Hb  = (unsigned short*)take(bb / 2);  // bf16 H
  unsigned short* HTb = (unsigned short*)take(bb / 2);  // bf16 H^T
  float* DE   = take(B);
  float* IDE  = take(B);
  float* DV   = take(B);
  float* DVIS = take(B);
  float* DV2  = take(B);
  float* T1   = take((size_t)B * 768);
  float* T2   = take((size_t)B * 768);
  float* TA   = take((size_t)B * 768);
  float* TB   = take((size_t)B * 768);
  float* X1   = take((size_t)B * 256);
  float* X2   = take((size_t)B * 128);
  float* G1   = take((size_t)B * 64);
  float* Rr   = take((size_t)B * 64);
  float* A1   = take((size_t)B * 32);
  float* D1   = take((size_t)B * 256);
  float* REC  = take((size_t)B * 768);
  float* NUM  = take(64);
  float* DEN  = take(64);
  float* RSUM = take(4);

  auto scale = [&](const float* src, const float* sc, float c, float* dst, int M, int Nc) {
    size_t tot = (size_t)M * Nc;
    scale_rows_kernel<<<(unsigned)((tot + 255) / 256), 256, 0, stream>>>(src, sc, c, dst, M, Nc);
  };
  // Lt@X = -(D1 H D2 H^T D1) X  (L = I - W, Lt = L - I = -W)
  auto ltx = [&](const float* X, int Nc, float* o, float* ta, float* tb) {
    scale(X, DVIS, 1.0f, ta, B, Nc);
    launch_gemm(stream, HTb, ta, nullptr, tb, B, Nc, B, Nc, Nc, 1.0f, 0, 0, 0, 0, 1);
    scale(tb, IDE, 1.0f, tb, B, Nc);
    launch_gemm(stream, Hb, tb, nullptr, ta, B, Nc, B, Nc, Nc, 1.0f, 0, 0, 0, 0, 1);
    scale(ta, DVIS, -1.0f, o, B, Nc);
  };

  // ---- init scratch that must be zeroed each call (bf16 0 == 0x0000)
  fill_kernel<<<(unsigned)((bb / 2 + 255) / 256), 256, 0, stream>>>((float*)Hb, 0.0f, bb / 2);
  fill_kernel<<<(unsigned)((bb / 2 + 255) / 256), 256, 0, stream>>>((float*)HTb, 0.0f, bb / 2);
  fill_kernel<<<(unsigned)((B + 255) / 256), 256, 0, stream>>>(DV, 1e-8f, (size_t)B);
  fill_kernel<<<(unsigned)((B + 255) / 256), 256, 0, stream>>>(DV2, 1e-8f, (size_t)B);
  fill_kernel<<<1, 256, 0, stream>>>(RSUM, 0.0f, 1);

  // ---- modality encoders
  pool_kernel<<<(unsigned)((B * 192 + 255) / 256), 256, 0, stream>>>(images, POOL, B);
  launch_gemm(stream, POOL, iw1, ib1, MH, B, 256, 192, 256, 256, 1.0f, 0, 1, 1, 0, 0);
  launch_gemm(stream, MH, iw2, ib2, MY, B, 256, 256, 256, 256, 1.0f, 0, 0, 1, 0, 0);
  bn_kernel<<<256, 256, 0, stream>>>(MY, ig, ibt, FE, B, 256, 768, 0);
  launch_gemm(stream, txt, tw1, tb1, MH, B, 256, 768, 256, 256, 1.0f, 0, 1, 1, 0, 0);
  launch_gemm(stream, MH, tw2, tb2, MY, B, 256, 256, 256, 256, 1.0f, 0, 0, 1, 0, 0);
  bn_kernel<<<256, 256, 0, stream>>>(MY, tg, tbt, FE, B, 256, 768, 256);
  launch_gemm(stream, sig, sw1, sb1, MH, B, 256, 256, 256, 256, 1.0f, 0, 1, 1, 0, 0);
  launch_gemm(stream, MH, sw2, sb2, MY, B, 256, 256, 256, 256, 1.0f, 0, 0, 1, 0, 0);
  bn_kernel<<<256, 256, 0, stream>>>(MY, sg, sbt, FE, B, 256, 768, 512);

  // ---- hyperedge generator: z = tanh(FE@gw+gb), row-normalize, sim = z z^T
  launch_gemm(stream, FE, gw, gb, Z, B, 256, 768, 256, 256, 1.0f, 0, 2, 1, 0, 0);
  rownorm_kernel<<<B, 256, 0, stream>>>(Z, 256);
  launch_gemm(stream, Z, Z, nullptr, SIM, B, B, 256, 256, B, 1.0f, 1, 0, 0, 0, 0);
  topk_kernel<<<B, 256, 0, stream>>>(SIM, Hb, HTb, DV, DV2, DE);
  prep_kernel<<<(unsigned)((B + 255) / 256), 256, 0, stream>>>(DV, DE, DVIS, IDE, B);

  const float e1 = 0.60653065971f;  // exp(-0.5)
  const float e2 = 0.36787944117f;  // exp(-1.0)

  // ---- Chebyshev conv 1: feats[3072,768] -> X1[3072,256], relu
  launch_gemm(stream, FE, c1w, nullptr, X1, B, 256, 768, 256, 256, 1.0f, 0, 0, 0, 0, 0);
  ltx(FE, 768, T1, TA, TB);
  launch_gemm(stream, T1, c1w + 768 * 256, nullptr, X1, B, 256, 768, 256, 256, e1, 0, 0, 0, 1, 0);
  ltx(T1, 768, T2, TA, TB);
  axpby_kernel<<<(unsigned)(((size_t)B * 768 + 255) / 256), 256, 0, stream>>>(
      2.0f, T2, -1.0f, FE, T2, (size_t)B * 768);
  launch_gemm(stream, T2, c1w + 2 * 768 * 256, c1b, X1, B, 256, 768, 256, 256, e2, 0, 1, 1, 1, 0);

  // ---- Chebyshev conv 2: X1[3072,256] -> X2[3072,128], relu
  launch_gemm(stream, X1, c2w, nullptr, X2, B, 128, 256, 128, 128, 1.0f, 0, 0, 0, 0, 0);
  ltx(X1, 256, T1, TA, TB);
  launch_gemm(stream, T1, c2w + 256 * 128, nullptr, X2, B, 128, 256, 128, 128, e1, 0, 0, 0, 1, 0);
  ltx(T1, 256, T2, TA, TB);
  axpby_kernel<<<(unsigned)(((size_t)B * 256 + 255) / 256), 256, 0, stream>>>(
      2.0f, T2, -1.0f, X1, T2, (size_t)B * 256);
  launch_gemm(stream, T2, c2w + 2 * 256 * 128, c2b, X2, B, 128, 256, 128, 128, e2, 0, 1, 1, 1, 0);

  // ---- heads
  launch_gemm(stream, X2, rw1, rb1, G1, B, 64, 128, 64, 64, 1.0f, 0, 1, 1, 0, 0);
  launch_gemm(stream, G1, rw2, rb2, Rr, B, 64, 64, 64, 64, 1.0f, 0, 0, 1, 0, 0);
  launch_gemm(stream, Rr, aw1, ab1, A1, B, 32, 64, 32, 32, 1.0f, 0, 1, 1, 0, 0);
  launch_gemm(stream, A1, aw2, ab2, out, B, 1, 32, 1, 1, 1.0f, 0, 3, 1, 0, 0);  // scores
  launch_gemm(stream, Rr, dw1, db1, D1, B, 256, 64, 256, 256, 1.0f, 0, 1, 1, 0, 0);
  launch_gemm(stream, D1, dw2, db2, REC, B, 768, 256, 768, 768, 1.0f, 0, 0, 1, 0, 0);
  sqdiff_kernel<<<(unsigned)(((size_t)B * 768 + 255) / 256), 256, 0, stream>>>(
      REC, FE, RSUM, (size_t)B * 768);

  // ---- spectral cut regularizer: t2 = (H/De) @ (H^T @ r)
  launch_gemm(stream, HTb, Rr, nullptr, TA, B, 64, B, 64, 64, 1.0f, 0, 0, 0, 0, 1);
  scale(TA, IDE, 1.0f, TA, B, 64);
  launch_gemm(stream, Hb, TA, nullptr, TB, B, 64, B, 64, 64, 1.0f, 0, 0, 0, 0, 1);
  speccols_kernel<<<64, 256, 0, stream>>>(Rr, TB, DV2, NUM, DEN, B, 64);

  finalize_kernel<<<1, 64, 0, stream>>>(RSUM, NUM, DEN, out + B,
                                        1.0f / ((float)B * 768.0f), 64);
}
